// InstCriterion_91293824843907
// MI455X (gfx1250) — compile-verified, matrix operand order checked by reading, not run on hardware
//
#include <hip/hip_runtime.h>

typedef float v2f __attribute__((ext_vector_type(2)));
typedef float v8f __attribute__((ext_vector_type(8)));
typedef _Float16 v16h __attribute__((ext_vector_type(16)));

constexpr int NPTS  = 40000;
constexpr int NMASK = 30000;
constexpr int NINST = 1024;
constexpr int NB    = 4;
constexpr int NLAB  = 42;   // instance label + 1 in [0, 41]
constexpr int NSEM  = 20;
constexpr int NLAY  = 4;    // layers 2..5

// ---- workspace layout (ints) ----
constexpr int OFF_DEN  = 0;                       // NB*NLAB   den histogram (all points)
constexpr int OFF_FGI  = OFF_DEN  + NB * NLAB;    // NB*NLAB   fg instance histogram
constexpr int OFF_SEM  = OFF_FGI  + NB * NLAB;    // NB*NSEM   fg semantic histogram
constexpr int OFF_JNT  = OFF_SEM  + NB * NSEM;    // NB*NSEM*NLAB joint (sem,inst) histogram
constexpr int OFF_BMIN = OFF_JNT  + NB * NSEM * NLAB;
constexpr int OFF_BMAX = OFF_BMIN + NB;
constexpr int OFF_RCNT = OFF_BMAX + NB;           // per-row: counted flag
constexpr int OFF_RWRW = OFF_RCNT + NINST;        // per-row: w_row
constexpr int OFF_RWGT = OFF_RWRW + NINST;        // per-row: sum(w*gt)
constexpr int OFF_RSEG = OFF_RWGT + NINST;        // per-row: s_seg
constexpr int OFF_RINS = OFF_RSEG + NINST;        // per-row: s_ins + 1
constexpr int OFF_END_I = OFF_RINS + NINST;
constexpr size_t KEYS_OFF   = ((size_t)OFF_END_I * 4 + 63) & ~(size_t)63;    // u16[NMASK]
constexpr size_t ROWOUT_OFF = (KEYS_OFF + (size_t)NMASK * 2 + 15) & ~(size_t)15; // float4[NLAY*NINST]

// --------------------------------------------------------------------------
// WMMA-based exact sum reduction (wave32). With B == all-ones,
// D[m][n] = sum_k A[m][k] + C[m][n]; summing every element of C across all
// lanes/VGPRs gives 16 * (sum of everything ever fed into A), regardless of
// the precise lane<->element mapping. So this is layout-agnostic and exact
// for the f32 variant.
// --------------------------------------------------------------------------
__device__ __forceinline__ void wmma_feed(v8f& c, float a0, float a1) {
#if __has_builtin(__builtin_amdgcn_wmma_f32_16x16x4_f32)
  v2f a; a[0] = a0; a[1] = a1;
  v2f b; b[0] = 1.0f; b[1] = 1.0f;
  c = __builtin_amdgcn_wmma_f32_16x16x4_f32(false, a, false, b, (short)0, c,
                                            false, false);
#else
  v16h a = {};
  a[0] = (_Float16)a0; a[1] = (_Float16)a1;
  v16h b;
#pragma unroll
  for (int i = 0; i < 16; ++i) b[i] = (_Float16)1.0f;
  c = __builtin_amdgcn_wmma_f32_16x16x32_f16(false, a, false, b, (short)0, c,
                                             false, false);
#endif
}

__device__ __forceinline__ float wmma_total(v8f c) {
  float s = 0.f;
#pragma unroll
  for (int i = 0; i < 8; ++i) s += c[i];
  for (int m = 16; m >= 1; m >>= 1) s += __shfl_xor(s, m, 32);
  return s * 0.0625f;  // /16: each of the 16 columns holds the full row-sums
}

// --------------------------------------------------------------------------
__global__ void k_init(int* wsI) {
  int t = blockIdx.x * blockDim.x + threadIdx.x;
  if (t >= OFF_RCNT) return;
  if (t >= OFF_BMAX)      wsI[t] = -1;      // bmax init
  else if (t >= OFF_BMIN) wsI[t] = NMASK;   // bmin init
  else                    wsI[t] = 0;       // histograms
}

__global__ void k_hist_pts(const int* __restrict__ instl,
                           const int* __restrict__ offs, int* wsI) {
  int p = blockIdx.x * blockDim.x + threadIdx.x;
  if (p >= NPTS) return;
  int cnt = 0;
#pragma unroll
  for (int k = 0; k < NB + 1; ++k) cnt += (offs[k] <= p) ? 1 : 0;
  int b = cnt - 1;
  if (b < 0) b = 0; if (b > NB - 1) b = NB - 1;
  atomicAdd(wsI + OFF_DEN + b * NLAB + (instl[p] + 1), 1);
}

__global__ void k_hist_fg(const int* __restrict__ fgidx,
                          const int* __restrict__ bidx,
                          const int* __restrict__ instl,
                          const int* __restrict__ seml, int* wsI,
                          unsigned short* __restrict__ keys) {
  int j = blockIdx.x * blockDim.x + threadIdx.x;
  if (j >= NMASK) return;
  int idx = fgidx[j];
  int b = bidx[j] & (NB - 1);
  int iv = instl[idx] + 1;   // [0,41)
  int sv = seml[idx];        // [0,20)
  atomicAdd(wsI + OFF_FGI + b * NLAB + iv, 1);
  atomicAdd(wsI + OFF_SEM + b * NSEM + sv, 1);
  atomicAdd(wsI + OFF_JNT + (b * NSEM + sv) * NLAB + iv, 1);
  atomicMin(wsI + OFF_BMIN + b, j);
  atomicMax(wsI + OFF_BMAX + b, j);
  keys[j] = (unsigned short)((sv << 6) | iv);
}

__global__ void k_rows(const int* __restrict__ fps,
                       const int* __restrict__ instl,
                       const int* __restrict__ seml, int* wsI) {
  int i = blockIdx.x * blockDim.x + threadIdx.x;
  if (i >= NINST) return;
  int b = i >> 8;                  // query batch: i // 256
  int fi = fps[i];
  int si = instl[fi];
  int ss = seml[fi];
  int num = wsI[OFF_FGI + b * NLAB + si + 1];
  int den = wsI[OFF_DEN + b * NLAB + si + 1];
  float cover = (float)num / (float)den;   // 0/0 -> NaN -> not positive (matches ref)
  bool positive = (si >= 0) && (ss >= 4) && (cover >= 0.3f);
  int wr = positive ? wsI[OFF_SEM + b * NSEM + ss] : 0;
  bool counted = positive && (wr > 0);
  int wg = counted ? wsI[OFF_JNT + (b * NSEM + ss) * NLAB + si + 1] : 0;
  wsI[OFF_RCNT + i] = counted ? 1 : 0;
  wsI[OFF_RWRW + i] = wr;
  wsI[OFF_RWGT + i] = wg;
  wsI[OFF_RSEG + i] = ss;
  wsI[OFF_RINS + i] = si + 1;
}

// One wave per (layer, row). 8 waves/block -> 8 consecutive rows, all in the
// same batch, so the block stages that batch's packed (sem,inst) keys in LDS
// once. Only lanes whose key passes the w-mask (~1/20) touch the logits.
__global__ void __launch_bounds__(256) k_main(
    const float* __restrict__ logits, const unsigned short* __restrict__ keys,
    const int* __restrict__ wsI, float4* __restrict__ rowOut) {
  __shared__ unsigned short lkeys[NMASK];
  int blk = blockIdx.x;
  int l  = blk >> 7;              // 0..3 -> layer l+2
  int r0 = (blk & 127) << 3;      // first of 8 rows
  int b  = r0 >> 8;               // batch of these rows
  int start = wsI[OFF_BMIN + b];
  int len = wsI[OFF_BMAX + b] + 1 - start;
  if (len < 0) len = 0;
  if (len > NMASK) len = NMASK;
  for (int t = threadIdx.x; t < len; t += 256) lkeys[t] = keys[start + t];
  __syncthreads();

  int lane = threadIdx.x & 31;
  int r = r0 + (threadIdx.x >> 5);
  float inter = 0.f, wp2 = 0.f, bce = 0.f;
  if (wsI[OFF_RCNT + r]) {
    int ss  = wsI[OFF_RSEG + r];
    int si1 = wsI[OFF_RINS + r];
    const float* rowp =
        logits + (size_t)((l + 2) * NINST + r) * NMASK + start;
    for (int t = lane; t < len; t += 32) {
      if ((lane & 7) == 0 && t + 256 < len)
        __builtin_prefetch(rowp + t + 256, 0, 0);   // global_prefetch_b8
      unsigned k = lkeys[t];
      if ((int)(k >> 6) == ss) {                    // w-mask (sem match)
        float x = rowp[t];
        float ax  = fabsf(x);
        float e   = __expf(-ax);                    // v_exp_f32
        float inv = __builtin_amdgcn_rcpf(1.0f + e);
        float pred = (x >= 0.f ? 1.0f : e) * inv;   // sigmoid(x)
        float sp   = fmaxf(x, 0.f) + __logf(1.0f + e); // softplus(x)
        bool g = (int)(k & 63u) == si1;             // gt (inst match)
        bce   += sp - (g ? x : 0.f);                // BCE = softplus(x) - g*x
        wp2   += pred * pred;
        inter += g ? pred : 0.f;
      }
    }
    for (int m = 16; m >= 1; m >>= 1) {
      inter += __shfl_xor(inter, m, 32);
      wp2   += __shfl_xor(wp2,   m, 32);
      bce   += __shfl_xor(bce,   m, 32);
    }
  }
  if (lane == 0) rowOut[l * NINST + r] = make_float4(inter, wp2, bce, 0.f);
}

__global__ void k_final(const float4* __restrict__ rowOut,
                        const int* __restrict__ wsI, float* __restrict__ out) {
  int lane = threadIdx.x;  // single wave of 32
  v8f cC = {}, cW = {};
  for (int base = 0; base < NINST; base += 64) {
    int i0 = base + lane, i1 = base + 32 + lane;
    wmma_feed(cC, (float)wsI[OFF_RCNT + i0], (float)wsI[OFF_RCNT + i1]);
    wmma_feed(cW, (float)wsI[OFF_RWRW + i0], (float)wsI[OFF_RWRW + i1]);
  }
  float csum = wmma_total(cC);   // sum(counted)
  float wsum = wmma_total(cW);   // sum(w)
  float loss = 0.f;
  for (int l = 0; l < NLAY; ++l) {
    v8f cD = {}, cB = {};
    for (int base = 0; base < NINST; base += 64) {
      float d[2], bb[2];
#pragma unroll
      for (int u = 0; u < 2; ++u) {
        int i = base + u * 32 + lane;
        float di = 0.f, bi = 0.f;
        if (wsI[OFF_RCNT + i]) {
          float4 ro = rowOut[l * NINST + i];
          float uni = ro.y + (float)wsI[OFF_RWGT + i] + 1e-5f;
          di = 1.0f - 2.0f * ro.x / uni;  // dice for counted row
          bi = ro.z;                      // masked BCE sum for row
        }
        d[u] = di; bb[u] = bi;
      }
      wmma_feed(cD, d[0], d[1]);
      wmma_feed(cB, bb[0], bb[1]);
    }
    float dsum = wmma_total(cD);
    float bsum = wmma_total(cB);
    loss += bsum / (wsum + 1e-6f) + dsum / (csum + 1e-6f);
  }
  if (lane == 0) out[0] = loss;
}

// --------------------------------------------------------------------------
extern "C" void kernel_launch(void* const* d_in, const int* in_sizes, int n_in,
                              void* d_out, int out_size, void* d_ws,
                              size_t ws_size, hipStream_t stream) {
  (void)in_sizes; (void)n_in; (void)out_size; (void)ws_size;
  const float* logits = (const float*)d_in[0];
  const int* seml  = (const int*)d_in[1];
  const int* instl = (const int*)d_in[2];
  const int* fgidx = (const int*)d_in[3];
  const int* bidx  = (const int*)d_in[4];
  const int* offs  = (const int*)d_in[5];
  const int* fps   = (const int*)d_in[6];
  float* out = (float*)d_out;

  int* wsI = (int*)d_ws;
  char* wsB = (char*)d_ws;
  unsigned short* keys = (unsigned short*)(wsB + KEYS_OFF);
  float4* rowOut = (float4*)(wsB + ROWOUT_OFF);

  k_init<<<(OFF_RCNT + 255) / 256, 256, 0, stream>>>(wsI);
  k_hist_pts<<<(NPTS + 255) / 256, 256, 0, stream>>>(instl, offs, wsI);
  k_hist_fg<<<(NMASK + 255) / 256, 256, 0, stream>>>(fgidx, bidx, instl, seml,
                                                     wsI, keys);
  k_rows<<<(NINST + 255) / 256, 256, 0, stream>>>(fps, instl, seml, wsI);
  k_main<<<NLAY * 128, 256, 0, stream>>>(logits, keys, wsI, rowOut);
  k_final<<<1, 32, 0, stream>>>(rowOut, wsI, out);
}
